// E_GCL_15135464751164
// MI455X (gfx1250) — matrix operand order repaired and checked by reading, not compiled, not run on hardware
//
#include <hip/hip_runtime.h>
#include <cstdint>

#define N_NODES 50000
#define N_EDGES 1600000
#define FDIM 64
#define HDIM 128
#define CLIPV 1000.0f

typedef _Float16 half8 __attribute__((ext_vector_type(8)));
typedef _Float16 v16h  __attribute__((ext_vector_type(16)));
typedef float    v8f   __attribute__((ext_vector_type(8)));
typedef unsigned int u32x4 __attribute__((ext_vector_type(4)));
typedef int          i32x8 __attribute__((ext_vector_type(8)));
typedef int          i32x4 __attribute__((ext_vector_type(4)));

#if defined(__gfx1250__) && __has_builtin(__builtin_amdgcn_tensor_load_to_lds) && __has_builtin(__builtin_amdgcn_s_wait_tensorcnt)
#define HAVE_TDM 1
#else
#define HAVE_TDM 0
#endif

// ---------------- device helpers ----------------

__device__ __forceinline__ float siluf(float v) {
    return v * (1.0f / (1.0f + __expf(-v)));
}

__device__ __forceinline__ half8 pack8(float4 a, float4 b) {
    half8 h;
    h[0] = (_Float16)a.x; h[1] = (_Float16)a.y; h[2] = (_Float16)a.z; h[3] = (_Float16)a.w;
    h[4] = (_Float16)b.x; h[5] = (_Float16)b.y; h[6] = (_Float16)b.z; h[7] = (_Float16)b.w;
    return h;
}

// A fragment: 16x32 f16, row M = lane%16, K = c*32 + half*16 + {0..15}.
__device__ __forceinline__ v16h ldA(const _Float16* base, int strideHalves, int c,
                                    int l16, int hf) {
    const _Float16* p = base + l16 * strideHalves + c * 32 + hf * 16;
    half8 lo = *(const half8*)p;
    half8 hi = *(const half8*)(p + 8);
    v16h a;
#pragma unroll
    for (int i = 0; i < 8; ++i) { a[i] = lo[i]; a[8 + i] = hi[i]; }
    return a;
}

// B fragment from transposed weights WT[n][k]: lane holds col n = nt*16+lane%16,
// K = c*32 + half*16 + {0..15} -> 32 contiguous bytes.
__device__ __forceinline__ v16h ldB(const _Float16* wt, int strideHalves, int nt, int c,
                                    int l16, int hf) {
    const _Float16* p = wt + (nt * 16 + l16) * strideHalves + c * 32 + hf * 16;
    return *(const v16h*)p;
}

__device__ __forceinline__ v8f wmma_f16(v16h a, v16h b, v8f c) {
    return __builtin_amdgcn_wmma_f32_16x16x32_f16(false, a, false, b, (short)0, c,
                                                  false, false);
}

#if HAVE_TDM
__device__ __forceinline__ unsigned lds_off(const void* p) {
    return (unsigned)(unsigned long long)(const __attribute__((address_space(3))) char*)p;
}
// Contiguous copy of nElems f16 from global to LDS via the Tensor Data Mover.
__device__ __forceinline__ void tdm_copy(unsigned ldsOff, const _Float16* g, unsigned nElems) {
    unsigned long long ga = (unsigned long long)(uintptr_t)g;
    u32x4 g0;
    g0[0] = 1u;                                     // count=1 valid descriptor
    g0[1] = ldsOff;                                 // lds_addr (bytes)
    g0[2] = (unsigned)ga;                           // global_addr[31:0]
    g0[3] = (unsigned)((ga >> 32) & 0x1FFFFFFull) | (2u << 30);  // addr hi | type=2
    i32x8 g1;
    g1[0] = (int)(1u << 16);                        // data_size=1 (2 bytes)
    g1[1] = (int)((nElems & 0xFFFFu) << 16);        // tensor_dim0[15:0]
    g1[2] = (int)(((nElems >> 16) & 0xFFFFu) | (1u << 16));  // dim0 hi | tensor_dim1=1
    g1[3] = (int)((nElems & 0xFFFFu) << 16);        // tile_dim0 = nElems (<=65535)
    g1[4] = 1;                                      // tile_dim1=1, tile_dim2=0
    g1[5] = (int)nElems;                            // tensor_dim0_stride lo
    g1[6] = 0;
    g1[7] = 0;
    i32x4 z4 = {0, 0, 0, 0};
    i32x8 z8 = {0, 0, 0, 0, 0, 0, 0, 0};
    __builtin_amdgcn_tensor_load_to_lds(g0, g1, z4, z4, z8, 0);
}
#endif

// ---------------- weight prep: fp32 -> transposed f16 ----------------
// hw layout (halves): W1bT[128][128] @0 | W2T[128][128] @16384 | Wc1T[128][128] @32768
//                     Wn1T[128][192] @49152 | Wn2T[64][128] @73728
__global__ void egcl_prep_kernel(const float* __restrict__ W1, const float* __restrict__ W2,
                                 const float* __restrict__ Wc1, const float* __restrict__ Wn1,
                                 const float* __restrict__ Wn2, _Float16* __restrict__ hw) {
    int idx = blockIdx.x * blockDim.x + threadIdx.x;
    if (idx >= 81920) return;
    float v;
    if (idx < 16384) {            // W1 body (rows 1..128), transposed
        int n = idx >> 7, k = idx & 127;
        v = W1[(k + 1) * HDIM + n];
    } else if (idx < 32768) {
        int j = idx - 16384; int n = j >> 7, k = j & 127;
        v = W2[k * HDIM + n];
    } else if (idx < 49152) {
        int j = idx - 32768; int n = j >> 7, k = j & 127;
        v = Wc1[k * HDIM + n];
    } else if (idx < 73728) {
        int j = idx - 49152; int n = j / 192, k = j % 192;
        v = Wn1[k * HDIM + n];
    } else {
        int j = idx - 73728; int n = j >> 7, k = j & 127;
        v = Wn2[k * FDIM + n];
    }
    hw[idx] = (_Float16)v;
}

// ---------------- edge kernel: fused 3-layer edge MLP + aggregation ----------------
__global__ __launch_bounds__(128) void egcl_edge_kernel(
    const float* __restrict__ x, const float* __restrict__ hh,
    const int* __restrict__ src, const int* __restrict__ dst,
    const float* __restrict__ W1, const float* __restrict__ b1,
    const float* __restrict__ b2, const float* __restrict__ bc1,
    const float* __restrict__ Wc2, const _Float16* __restrict__ hw,
    float* __restrict__ ef_sum, float* __restrict__ trans_sum,
    float* __restrict__ counts) {
    __shared__ _Float16 sW[3 * 128 * 128];       // W1bT | W2T | Wc1T  (96 KB)
    __shared__ _Float16 sBuf[4][2][16 * 128];    // per-wave ping/pong activation tiles
    __shared__ float sRad[4][16];
    __shared__ float sCv[4][16];
    __shared__ float sCd[4][16][3];
    __shared__ int sDst[4][16];

    const int tid = threadIdx.x;
    const int wv = tid >> 5;
    const int lane = tid & 31;
    const int l16 = lane & 15;
    const int hf = lane >> 4;

#if HAVE_TDM
    if (wv == 0) {
        tdm_copy(lds_off(sW), hw, 49152);        // all three matrices, contiguous
        __builtin_amdgcn_s_wait_tensorcnt(0);
    }
#else
    {
        const half8* g = (const half8*)hw;
        half8* s = (half8*)sW;
        for (int i = tid; i < 49152 / 8; i += 128) s[i] = g[i];
    }
#endif
    __syncthreads();

    // per-lane column constants (col = nt*16 + l16)
    float bb1[8], bb2[8], bbc1[8], w1r0[8], wc2c[8];
#pragma unroll
    for (int nt = 0; nt < 8; ++nt) {
        int col = nt * 16 + l16;
        bb1[nt] = b1[col]; bb2[nt] = b2[col]; bbc1[nt] = bc1[col];
        w1r0[nt] = W1[col];                      // W1 row 0 = radial weights
        wc2c[nt] = Wc2[col];
    }

    const v8f vzero = {0.f, 0.f, 0.f, 0.f, 0.f, 0.f, 0.f, 0.f};
    const int numTiles = N_EDGES / 16;
    const int wave0 = blockIdx.x * 4 + wv;
    const int waveStride = gridDim.x * 4;

    for (int tile = wave0; tile < numTiles; tile += waveStride) {
        const int e0 = tile * 16;
        if (hf == 0) {
            int s_ = src[e0 + l16], d_ = dst[e0 + l16];
            float c0 = x[s_ * 3 + 0] - x[d_ * 3 + 0];
            float c1 = x[s_ * 3 + 1] - x[d_ * 3 + 1];
            float c2 = x[s_ * 3 + 2] - x[d_ * 3 + 2];
            sDst[wv][l16] = d_;
            sCd[wv][l16][0] = c0; sCd[wv][l16][1] = c1; sCd[wv][l16][2] = c2;
            sRad[wv][l16] = c0 * c0 + c1 * c1 + c2 * c2;
        }
        {   // gather [hh[src] | hh[dst]] -> 16x128 f16 A tile (buf0)
            int idx = hf ? dst[e0 + l16] : src[e0 + l16];
            const float4* hp = (const float4*)(hh + (size_t)idx * FDIM);
            half8* row = (half8*)&sBuf[wv][0][l16 * 128 + hf * 64];
#pragma unroll
            for (int i = 0; i < 8; ++i) row[i] = pack8(hp[2 * i], hp[2 * i + 1]);
        }
        asm volatile("" ::: "memory");

        v8f acc[8];
        // ---- layer 1: radial rank-1 init + WMMA over K=128 ----
        float rad[8];
#pragma unroll
        for (int v = 0; v < 8; ++v) rad[v] = sRad[wv][hf * 8 + v];
#pragma unroll
        for (int nt = 0; nt < 8; ++nt) {
#pragma unroll
            for (int v = 0; v < 8; ++v) acc[nt][v] = rad[v] * w1r0[nt];
        }
#pragma unroll
        for (int c = 0; c < 4; ++c) {
            v16h a = ldA(&sBuf[wv][0][0], 128, c, l16, hf);
#pragma unroll
            for (int nt = 0; nt < 8; ++nt)
                acc[nt] = wmma_f16(a, ldB(&sW[0], 128, nt, c, l16, hf), acc[nt]);
        }
#pragma unroll
        for (int nt = 0; nt < 8; ++nt) {
            int col = nt * 16 + l16;
#pragma unroll
            for (int v = 0; v < 8; ++v)
                sBuf[wv][1][(v + 8 * hf) * 128 + col] = (_Float16)siluf(acc[nt][v] + bb1[nt]);
        }
        asm volatile("" ::: "memory");

        // ---- layer 2 ----
#pragma unroll
        for (int nt = 0; nt < 8; ++nt) acc[nt] = vzero;
#pragma unroll
        for (int c = 0; c < 4; ++c) {
            v16h a = ldA(&sBuf[wv][1][0], 128, c, l16, hf);
#pragma unroll
            for (int nt = 0; nt < 8; ++nt)
                acc[nt] = wmma_f16(a, ldB(&sW[16384], 128, nt, c, l16, hf), acc[nt]);
        }
        // ef = silu(.): store act for layer3 + run-compressed atomic aggregation (dst sorted)
        int dv[8];
#pragma unroll
        for (int v = 0; v < 8; ++v) dv[v] = sDst[wv][hf * 8 + v];
#pragma unroll
        for (int nt = 0; nt < 8; ++nt) {
            int col = nt * 16 + l16;
            float run = 0.f; int cur = dv[0];
#pragma unroll
            for (int v = 0; v < 8; ++v) {
                float t = siluf(acc[nt][v] + bb2[nt]);
                sBuf[wv][0][(v + 8 * hf) * 128 + col] = (_Float16)t;
                if (dv[v] != cur) {
                    atomicAdd(&ef_sum[(size_t)cur * HDIM + col], run);
                    run = 0.f; cur = dv[v];
                }
                run += t;
            }
            atomicAdd(&ef_sum[(size_t)cur * HDIM + col], run);
        }
        asm volatile("" ::: "memory");

        // ---- layer 3 (coord MLP hidden) + Wc2 dot ----
#pragma unroll
        for (int nt = 0; nt < 8; ++nt) acc[nt] = vzero;
#pragma unroll
        for (int c = 0; c < 4; ++c) {
            v16h a = ldA(&sBuf[wv][0][0], 128, c, l16, hf);
#pragma unroll
            for (int nt = 0; nt < 8; ++nt)
                acc[nt] = wmma_f16(a, ldB(&sW[32768], 128, nt, c, l16, hf), acc[nt]);
        }
        float part[8];
#pragma unroll
        for (int v = 0; v < 8; ++v) part[v] = 0.f;
#pragma unroll
        for (int nt = 0; nt < 8; ++nt) {
#pragma unroll
            for (int v = 0; v < 8; ++v)
                part[v] += siluf(acc[nt][v] + bbc1[nt]) * wc2c[nt];
        }
#pragma unroll
        for (int m = 1; m < 16; m <<= 1) {
#pragma unroll
            for (int v = 0; v < 8; ++v) part[v] += __shfl_xor(part[v], m, 32);
        }
        if (l16 == 0) {
#pragma unroll
            for (int v = 0; v < 8; ++v) sCv[wv][hf * 8 + v] = part[v];
        }
        asm volatile("" ::: "memory");
        if (hf == 0) {
            int d_ = sDst[wv][l16];
            float cv = sCv[wv][l16];
#pragma unroll
            for (int k = 0; k < 3; ++k) {
                float t = sCd[wv][l16][k] * cv;
                t = fminf(fmaxf(t, -CLIPV), CLIPV);
                atomicAdd(&trans_sum[(size_t)d_ * 3 + k], t);
            }
            atomicAdd(&counts[d_], 1.0f);
        }
    }
}

// ---------------- node kernel: node MLP + residual + coord epilogue ----------------
__global__ __launch_bounds__(128) void egcl_node_kernel(
    const float* __restrict__ x, const float* __restrict__ hh,
    const float* __restrict__ bn1, const float* __restrict__ bn2,
    const _Float16* __restrict__ wn1t, const _Float16* __restrict__ wn2t,
    const float* __restrict__ ef_sum, const float* __restrict__ trans_sum,
    const float* __restrict__ counts, float* __restrict__ outC,
    float* __restrict__ outH) {
    __shared__ _Float16 sW1[128 * 192];          // Wn1T (48 KB)
    __shared__ _Float16 sW2[64 * 128];           // Wn2T (16 KB)
    __shared__ _Float16 sAgg[4][16 * 192];
    __shared__ _Float16 sAct[4][16 * 128];

    const int tid = threadIdx.x;
    const int wv = tid >> 5;
    const int lane = tid & 31;
    const int l16 = lane & 15;
    const int hf = lane >> 4;

#if HAVE_TDM
    if (wv == 0) {
        tdm_copy(lds_off(sW1), wn1t, 24576);
        tdm_copy(lds_off(sW2), wn2t, 8192);
        __builtin_amdgcn_s_wait_tensorcnt(0);
    }
#else
    {
        const half8* g1 = (const half8*)wn1t;
        half8* s1 = (half8*)sW1;
        for (int i = tid; i < 24576 / 8; i += 128) s1[i] = g1[i];
        const half8* g2 = (const half8*)wn2t;
        half8* s2 = (half8*)sW2;
        for (int i = tid; i < 8192 / 8; i += 128) s2[i] = g2[i];
    }
#endif
    __syncthreads();

    float bbn1[8];
#pragma unroll
    for (int nt = 0; nt < 8; ++nt) bbn1[nt] = bn1[nt * 16 + l16];
    float bbn2[4];
#pragma unroll
    for (int nt = 0; nt < 4; ++nt) bbn2[nt] = bn2[nt * 16 + l16];

    const v8f vzero = {0.f, 0.f, 0.f, 0.f, 0.f, 0.f, 0.f, 0.f};
    const int numTiles = N_NODES / 16;
    const int wave0 = blockIdx.x * 4 + wv;
    const int waveStride = gridDim.x * 4;

    for (int tile = wave0; tile < numTiles; tile += waveStride) {
        const int n0 = tile * 16;
        {   // gather [hh | ef_sum] -> 16x192 f16 A tile
            int n = n0 + l16;
            _Float16* row = &sAgg[wv][l16 * 192];
            const float4* hp = (const float4*)(hh + (size_t)n * FDIM);
            const float4* ep = (const float4*)(ef_sum + (size_t)n * HDIM);
            if (hf == 0) {
#pragma unroll
                for (int i = 0; i < 8; ++i)
                    ((half8*)row)[i] = pack8(hp[2 * i], hp[2 * i + 1]);          // cols 0..63
#pragma unroll
                for (int i = 0; i < 4; ++i)
                    ((half8*)(row + 64))[i] = pack8(ep[2 * i], ep[2 * i + 1]);   // cols 64..95
            } else {
#pragma unroll
                for (int i = 0; i < 12; ++i)
                    ((half8*)(row + 96))[i] = pack8(ep[8 + 2 * i], ep[9 + 2 * i]); // 96..191
            }
        }
        asm volatile("" ::: "memory");

        v8f acc[8];
#pragma unroll
        for (int nt = 0; nt < 8; ++nt) acc[nt] = vzero;
#pragma unroll
        for (int c = 0; c < 6; ++c) {
            v16h a = ldA(&sAgg[wv][0], 192, c, l16, hf);
#pragma unroll
            for (int nt = 0; nt < 8; ++nt)
                acc[nt] = wmma_f16(a, ldB(sW1, 192, nt, c, l16, hf), acc[nt]);
        }
#pragma unroll
        for (int nt = 0; nt < 8; ++nt) {
            int col = nt * 16 + l16;
#pragma unroll
            for (int v = 0; v < 8; ++v)
                sAct[wv][(v + 8 * hf) * 128 + col] = (_Float16)siluf(acc[nt][v] + bbn1[nt]);
        }
        asm volatile("" ::: "memory");

        v8f acc2[4];
#pragma unroll
        for (int nt = 0; nt < 4; ++nt) acc2[nt] = vzero;
#pragma unroll
        for (int c = 0; c < 4; ++c) {
            v16h a = ldA(&sAct[wv][0], 128, c, l16, hf);
#pragma unroll
            for (int nt = 0; nt < 4; ++nt)
                acc2[nt] = wmma_f16(a, ldB(sW2, 128, nt, c, l16, hf), acc2[nt]);
        }
#pragma unroll
        for (int nt = 0; nt < 4; ++nt) {
            int col = nt * 16 + l16;
#pragma unroll
            for (int v = 0; v < 8; ++v) {
                int n = n0 + v + 8 * hf;
                outH[(size_t)n * FDIM + col] =
                    acc2[nt][v] + bbn2[nt] + hh[(size_t)n * FDIM + col];
            }
        }
        if (hf == 0) {
            int n = n0 + l16;
            float cnt = fmaxf(counts[n], 1.0f);
#pragma unroll
            for (int k = 0; k < 3; ++k) {
                float xc = fminf(fmaxf(x[n * 3 + k], -CLIPV), CLIPV);
                outC[n * 3 + k] = xc + trans_sum[n * 3 + k] / cnt;
            }
        }
    }
}

// ---------------- launcher ----------------
extern "C" void kernel_launch(void* const* d_in, const int* in_sizes, int n_in,
                              void* d_out, int out_size, void* d_ws, size_t ws_size,
                              hipStream_t stream) {
    (void)in_sizes; (void)n_in; (void)out_size; (void)ws_size;
    const float* x   = (const float*)d_in[0];
    const float* hh  = (const float*)d_in[1];
    const int*   src = (const int*)d_in[2];
    const int*   dst = (const int*)d_in[3];
    const float* W1  = (const float*)d_in[4];
    const float* b1  = (const float*)d_in[5];
    const float* W2  = (const float*)d_in[6];
    const float* b2  = (const float*)d_in[7];
    const float* Wc1 = (const float*)d_in[8];
    const float* bc1 = (const float*)d_in[9];
    const float* Wc2 = (const float*)d_in[10];
    const float* Wn1 = (const float*)d_in[11];
    const float* bn1 = (const float*)d_in[12];
    const float* Wn2 = (const float*)d_in[13];
    const float* bn2 = (const float*)d_in[14];

    // workspace layout
    char* ws = (char*)d_ws;
    const size_t EF_BYTES    = (size_t)N_NODES * HDIM * 4;  // 25,600,000
    const size_t TR_OFF      = EF_BYTES;                    // trans_sum: N*3 f32
    const size_t CNT_OFF     = TR_OFF + (size_t)N_NODES * 3 * 4;
    const size_t AGG_BYTES   = CNT_OFF + (size_t)N_NODES * 4;  // 26,400,000
    float*     ef_sum    = (float*)ws;
    float*     trans_sum = (float*)(ws + TR_OFF);
    float*     counts    = (float*)(ws + CNT_OFF);
    _Float16*  hw        = (_Float16*)(ws + AGG_BYTES);

    (void)hipMemsetAsync(d_ws, 0, AGG_BYTES, stream);

    egcl_prep_kernel<<<(81920 + 255) / 256, 256, 0, stream>>>(W1, W2, Wc1, Wn1, Wn2, hw);

    egcl_edge_kernel<<<1024, 128, 0, stream>>>(x, hh, src, dst, W1, b1, b2, bc1, Wc2, hw,
                                               ef_sum, trans_sum, counts);

    float* outC = (float*)d_out;
    float* outH = outC + (size_t)N_NODES * 3;
    egcl_node_kernel<<<256, 128, 0, stream>>>(x, hh, bn1, bn2, hw + 49152, hw + 73728,
                                              ef_sum, trans_sum, counts, outC, outH);
}
